// MultiHeadAttention_90486370992905
// MI455X (gfx1250) — matrix operand order repaired and checked by reading, not compiled, not run on hardware
//
#include <hip/hip_runtime.h>

typedef unsigned short u16t;
typedef __attribute__((ext_vector_type(16))) __bf16 v16bf;
typedef __attribute__((ext_vector_type(8)))  float  v8f;

struct B16x16 { uint4 lo, hi; };

// Load 16 bf16 values as two 16B chunks (A/B WMMA per-lane fragments).
static __device__ __forceinline__ v16bf ld_pair(const u16t* p0, const u16t* p1) {
  B16x16 t;
  t.lo = *(const uint4*)p0;
  t.hi = *(const uint4*)p1;
  return __builtin_bit_cast(v16bf, t);
}

static __device__ __forceinline__ u16t f2bf(float f) {
  unsigned u = __float_as_uint(f);
  u += 0x7FFFu + ((u >> 16) & 1u);
  return (u16t)(u >> 16);
}

static __device__ __forceinline__ v8f wmma_bf16(v16bf a, v16bf b, v8f c) {
  return __builtin_amdgcn_wmma_f32_16x16x32_bf16(false, a, false, b, (short)0, c,
                                                 false, false);
}

// Issue one async global->LDS 16B copy per lane (ASYNCcnt-tracked, no VGPR data).
static __device__ __forceinline__ void async_ld_b128(unsigned lds_off,
                                                     unsigned gbyte_off,
                                                     const u16t* base) {
  asm volatile("global_load_async_to_lds_b128 %0, %1, %2"
               :: "v"(lds_off), "v"(gbyte_off), "s"(base) : "memory");
}

// ---------------------------------------------------------------- fp32 -> bf16
__global__ void cvt_kernel(const float* __restrict__ s, u16t* __restrict__ d, int n) {
  int i = blockIdx.x * blockDim.x + threadIdx.x;
  if (i < n) d[i] = f2bf(s[i]);
}

// ---------------------------------------------------------------- NT GEMM
// D[m,n] = (sum_k A[m,k]*B[n,k] + bias[n]) * scale  (software-pipelined)
// MODE 0: bf16 [M,N]; MODE 1: bf16 transposed per batch (Vt); MODE 2: fp32 [M,N]
template <int MODE>
__global__ __launch_bounds__(256) void gemm_nt_kernel(
    const u16t* __restrict__ A, const u16t* __restrict__ B,
    const float* __restrict__ bias, void* __restrict__ out,
    int M, int N, int K, float scale) {
  const int lane = threadIdx.x & 31;
  const int wave = threadIdx.x >> 5;
  const int r    = lane & 15;
  const int ah   = (lane >> 4) & 1;

  const int tiles_n = N >> 6;
  const int t  = blockIdx.x * 8 + wave;
  const int tm = t / tiles_n;
  const int tn = t % tiles_n;

  const u16t* Arow  = A + (size_t)(tm * 16 + r) * K + ah * 16;
  const u16t* Brow0 = B + (size_t)(tn * 64 + r) * K + ah * 16;

  v8f acc[4];
  v8f zero = {};
#pragma unroll
  for (int s = 0; s < 4; ++s) acc[s] = zero;

  v16bf a = ld_pair(Arow, Arow + 8);
  v16bf b[4];
#pragma unroll
  for (int s = 0; s < 4; ++s) {
    const u16t* br = Brow0 + (size_t)(s * 16) * K;
    b[s] = ld_pair(br, br + 8);
  }

  for (int k = 0; k < K; k += 32) {
    const int kp = (k + 32 < K) ? k + 32 : 0;   // prefetch next chunk
    v16bf an = ld_pair(Arow + kp, Arow + kp + 8);
    v16bf bn[4];
#pragma unroll
    for (int s = 0; s < 4; ++s) {
      const u16t* br = Brow0 + (size_t)(s * 16) * K + kp;
      bn[s] = ld_pair(br, br + 8);
    }
#pragma unroll
    for (int s = 0; s < 4; ++s) acc[s] = wmma_bf16(a, b[s], acc[s]);
    a = an;
#pragma unroll
    for (int s = 0; s < 4; ++s) b[s] = bn[s];
  }

  const int rbase = tm * 16 + ah * 8;
#pragma unroll
  for (int s = 0; s < 4; ++s) {
    const int n = tn * 64 + s * 16 + r;
    const float bn = bias[n];
#pragma unroll
    for (int v = 0; v < 8; ++v) {
      const int m = rbase + v;
      const float val = (acc[s][v] + bn) * scale;
      if (MODE == 0) {
        ((u16t*)out)[(size_t)m * N + n] = f2bf(val);
      } else if (MODE == 1) {
        const int bb = m >> 11;
        const int sk = m & 2047;
        ((u16t*)out)[(size_t)bb * N * 2048 + (size_t)n * 2048 + sk] = f2bf(val);
      } else {
        ((float*)out)[(size_t)m * N + n] = val;
      }
    }
  }
}

// ---------------------------------------------------------------- fused flash attention
// Block = 4 waves, same (b,h), 4 consecutive q-tiles. K/V chunks staged in LDS
// via double-buffered async copies (fetched once per block, not once per wave);
// LDS fragment reads are register-double-buffered to hide ds latency.
__global__ __launch_bounds__(128) void flash_attn_kernel(
    const u16t* __restrict__ Q, const u16t* __restrict__ Km,
    const u16t* __restrict__ Vt, u16t* __restrict__ Ctx) {
  constexpr int S = 2048, E = 4096, HD = 512, NH = 8;

  __shared__ __align__(16) u16t Kbuf[2][32][520];   // 32 k-rows x 512 dd (+pad)
  __shared__ __align__(16) u16t Vbuf[2][512][40];   // 512 dd-rows x 32 k (+pad)
  __shared__ __align__(16) u16t Ptile[4][16 * 40];  // per-wave P staging

  const int lane = threadIdx.x & 31;
  const int wave = threadIdx.x >> 5;
  const int r    = lane & 15;
  const int ah   = (lane >> 4) & 1;

  const int bh = blockIdx.x >> 5;                 // 16 (b,h) pairs
  const int qt = (blockIdx.x & 31) * 4 + wave;    // 128 q-tiles per (b,h)
  const int bb = bh >> 3;
  const int hh = bh & 7;

  const u16t* Qb = Q  + (size_t)(bb * S) * E + hh * HD;
  const u16t* Kb = Km + (size_t)(bb * S) * E + hh * HD;
  const u16t* Vb = Vt + (size_t)(bb * NH + hh) * HD * S;
  u16t* myl = Ptile[wave];

  // ---- cooperative async chunk fetch (32 k-rows of K, 32 k-cols of V)
  const int tt   = threadIdx.x;           // 0..127
  const int kth  = tt >> 6;               // K: row parity (2 rows per step)
  const int ksub = (tt & 63) * 8;         // K: 8-half sub-offset within a row
  const int vdd  = tt >> 2;               // V: dd row (32 rows per step)
  const int vsub = (tt & 3) * 8;

  auto issue_chunk = [&](int k0, int buf) {
    unsigned kg = ((unsigned)(k0 + kth) * E + (unsigned)ksub) * 2u;
    unsigned kl = (unsigned)(uintptr_t)&Kbuf[buf][kth][ksub];
    unsigned vg = ((unsigned)vdd * S + (unsigned)(k0 + vsub)) * 2u;
    unsigned vl = (unsigned)(uintptr_t)&Vbuf[buf][vdd][vsub];
#pragma unroll
    for (int i = 0; i < 16; ++i) {
      async_ld_b128(kl, kg, Kb);
      async_ld_b128(vl, vg, Vb);
      kg += 2u * E * 2u;       kl += 2u * 520u * 2u;
      vg += 32u * S * 2u;      vl += 32u * 40u * 2u;
    }
  };

  // ---- persistent Q tile in A-layout (16 chunks of K=32 -> 128 VGPRs)
  v16bf qa[16];
  const u16t* qrow = Qb + (size_t)(qt * 16 + r) * E + ah * 16;
#pragma unroll
  for (int c = 0; c < 16; ++c)
    qa[c] = ld_pair(qrow + c * 32, qrow + c * 32 + 8);

  v8f acc[32];
  v8f zero = {};
#pragma unroll
  for (int nt = 0; nt < 32; ++nt) acc[nt] = zero;

  float rowmax[8], rowsum[8];
#pragma unroll
  for (int v = 0; v < 8; ++v) { rowmax[v] = -__builtin_inff(); rowsum[v] = 0.f; }

  issue_chunk(0, 0);

  for (int k0 = 0; k0 < S; k0 += 32) {
    const int buf = (k0 >> 5) & 1;
    if (k0 + 32 < S) {
      issue_chunk(k0 + 32, buf ^ 1);
      // in-order retirement: <=32 outstanding => current chunk is in LDS
      asm volatile("s_wait_asynccnt 0x20" ::: "memory");
    } else {
      asm volatile("s_wait_asynccnt 0x0" ::: "memory");
    }
    __syncthreads();

    // ---- scores: two 16x16 tiles from the shared K chunk (pipelined ds reads)
    v8f s0 = zero, s1 = zero;
    const u16t* kp0 = &Kbuf[buf][r][ah * 16];
    const u16t* kp1 = &Kbuf[buf][16 + r][ah * 16];
    v16bf kb0 = ld_pair(kp0, kp0 + 8);
    v16bf kb1 = ld_pair(kp1, kp1 + 8);
#pragma unroll
    for (int c = 0; c < 16; ++c) {
      const int cn = (c + 1) & 15;               // wrap: last prefetch is dead
      v16bf nb0 = ld_pair(kp0 + cn * 32, kp0 + cn * 32 + 8);
      v16bf nb1 = ld_pair(kp1 + cn * 32, kp1 + cn * 32 + 8);
      s0 = wmma_bf16(qa[c], kb0, s0);
      s1 = wmma_bf16(qa[c], kb1, s1);
      kb0 = nb0;
      kb1 = nb1;
    }

    // ---- online softmax (row = v + 8*ah; reduce across the 16-lane half)
    float alph[8];
#pragma unroll
    for (int v = 0; v < 8; ++v) {
      float mx = fmaxf(s0[v], s1[v]);
#pragma unroll
      for (int off = 1; off <= 8; off <<= 1)
        mx = fmaxf(mx, __shfl_xor(mx, off, 32));
      const float mnew  = fmaxf(rowmax[v], mx);
      const float alpha = __expf(rowmax[v] - mnew);
      rowmax[v] = mnew;
      const float p0 = __expf(s0[v] - mnew);
      const float p1 = __expf(s1[v] - mnew);
      float ps = p0 + p1;
#pragma unroll
      for (int off = 1; off <= 8; off <<= 1)
        ps += __shfl_xor(ps, off, 32);
      rowsum[v] = rowsum[v] * alpha + ps;
      alph[v] = alpha;
      const int row = v + 8 * ah;
      myl[row * 40 + r]      = f2bf(p0);
      myl[row * 40 + r + 16] = f2bf(p1);
    }

#pragma unroll
    for (int nt = 0; nt < 32; ++nt)
#pragma unroll
      for (int v = 0; v < 8; ++v) acc[nt][v] *= alph[v];

    // read P back in A-layout (DS ops are in-order within a wave)
    const u16t* prow = &myl[r * 40 + ah * 16];
    v16bf pa = ld_pair(prow, prow + 8);

    // ---- ctx += P @ V from the shared V chunk (pipelined ds reads)
    const u16t* vp0 = &Vbuf[buf][r][ah * 16];
    v16bf vb = ld_pair(vp0, vp0 + 8);
#pragma unroll
    for (int nt = 0; nt < 32; ++nt) {
      const int nn = (nt + 1) & 31;              // wrap: last prefetch is dead
      const u16t* vp = vp0 + (size_t)(nn * 16) * 40;
      v16bf nvb = ld_pair(vp, vp + 8);
      acc[nt] = wmma_bf16(pa, vb, acc[nt]);
      vb = nvb;
    }
    __syncthreads();   // all waves done reading before next chunk overwrites
  }

  // ---- normalize + store ctx (bf16, [b*s, e] row-major)
  float inv[8];
#pragma unroll
  for (int v = 0; v < 8; ++v) inv[v] = 1.0f / rowsum[v];

  u16t* cb = Ctx + (size_t)(bb * S + qt * 16) * E + hh * HD;
#pragma unroll
  for (int nt = 0; nt < 32; ++nt) {
    const int n = nt * 16 + r;
#pragma unroll
    for (int v = 0; v < 8; ++v)
      cb[(size_t)(v + 8 * ah) * E + n] = f2bf(acc[nt][v] * inv[v]);
  }
}

// ---------------------------------------------------------------- launch
extern "C" void kernel_launch(void* const* d_in, const int* in_sizes, int n_in,
                              void* d_out, int out_size, void* d_ws, size_t ws_size,
                              hipStream_t stream) {
  const float* x  = (const float*)d_in[0];
  const float* Wq = (const float*)d_in[1];
  const float* bq = (const float*)d_in[2];
  const float* Wk = (const float*)d_in[3];
  const float* bk = (const float*)d_in[4];
  const float* Wv = (const float*)d_in[5];
  const float* bv = (const float*)d_in[6];
  const float* Wo = (const float*)d_in[7];
  const float* bo = (const float*)d_in[8];

  constexpr int B = 2, S = 2048, D = 512, E = 4096;
  constexpr size_t NX = (size_t)B * S * D;
  constexpr size_t NW = (size_t)E * D;
  constexpr size_t NQ = (size_t)B * S * E;

  u16t* p   = (u16t*)d_ws;
  u16t* xb  = p;            p += NX;
  u16t* wqb = p;            p += NW;
  u16t* wkb = p;            p += NW;
  u16t* wvb = p;            p += NW;
  u16t* wob = p;            p += NW;
  u16t* Qb  = p;            p += NQ;
  u16t* Kb  = p;            p += NQ;
  u16t* Vtb = p;            p += NQ;
  u16t* Cxb = p;            p += NQ;

  cvt_kernel<<<(int)(NX / 256), 256, 0, stream>>>(x,  xb,  (int)NX);
  cvt_kernel<<<(int)(NW / 256), 256, 0, stream>>>(Wq, wqb, (int)NW);
  cvt_kernel<<<(int)(NW / 256), 256, 0, stream>>>(Wk, wkb, (int)NW);
  cvt_kernel<<<(int)(NW / 256), 256, 0, stream>>>(Wv, wvb, (int)NW);
  cvt_kernel<<<(int)(NW / 256), 256, 0, stream>>>(Wo, wob, (int)NW);

  const int M = B * S;
  const float qscale = 0.04419417382415922f;  // 1/sqrt(512)

  gemm_nt_kernel<0><<<2048, 256, 0, stream>>>(xb, wqb, bq, Qb,  M, E, D, qscale);
  gemm_nt_kernel<0><<<2048, 256, 0, stream>>>(xb, wkb, bk, Kb,  M, E, D, 1.0f);
  gemm_nt_kernel<1><<<2048, 256, 0, stream>>>(xb, wvb, bv, Vtb, M, E, D, 1.0f);

  flash_attn_kernel<<<512, 128, 0, stream>>>(Qb, Kb, Vtb, Cxb);

  gemm_nt_kernel<2><<<256, 256, 0, stream>>>(Cxb, wob, bo, d_out, M, D, E, 1.0f);
}
